// GMQA_with_KV_17325898072504
// MI455X (gfx1250) — matrix operand order repaired and checked by reading, not compiled, not run on hardware
//
#include <hip/hip_runtime.h>
#include <math.h>

typedef __attribute__((ext_vector_type(16))) __bf16 v16bf;
typedef __attribute__((ext_vector_type(8)))  float  v8f;
typedef __attribute__((ext_vector_type(4)))  unsigned int v4u;
typedef __attribute__((ext_vector_type(4)))  float  v4f;

union Frag16 { v16bf v; v4u u[2]; };

__device__ __forceinline__ unsigned short f2bf(float f) {
  __bf16 b = (__bf16)f;                       // native RNE convert
  return __builtin_bit_cast(unsigned short, b);
}

// ---- DPP 16-lane all-reduce (masks {1,2,7,15} form a GF(2) basis of 16 lanes)
#define DPPF(x, ctrl) __builtin_bit_cast(float,                                   \
    __builtin_amdgcn_mov_dpp(__builtin_bit_cast(int, (x)), (ctrl), 0xf, 0xf, true))

__device__ __forceinline__ float dpp_max16(float x) {
  x = fmaxf(x, DPPF(x, 0xB1));   // quad_perm(1,0,3,2)  : xor 1
  x = fmaxf(x, DPPF(x, 0x4E));   // quad_perm(2,3,0,1)  : xor 2
  x = fmaxf(x, DPPF(x, 0x141));  // row_half_mirror     : xor 7
  x = fmaxf(x, DPPF(x, 0x140));  // row_mirror          : xor 15
  return x;
}
__device__ __forceinline__ float dpp_sum16(float x) {
  x += DPPF(x, 0xB1);
  x += DPPF(x, 0x4E);
  x += DPPF(x, 0x141);
  x += DPPF(x, 0x140);
  return x;
}

// ======================= pre-pass: f32 -> bf16 (row-major) ====================
__global__ __launch_bounds__(256)
void conv_bf16(const float* __restrict__ X, unsigned short* __restrict__ Xb, long n)
{
  long i = ((long)blockIdx.x * 256 + threadIdx.x) * 4;
  if (i + 3 < n) {
    v4f v = *reinterpret_cast<const v4f*>(X + i);
    unsigned long long p = (unsigned long long)f2bf(v[0])
                         | ((unsigned long long)f2bf(v[1]) << 16)
                         | ((unsigned long long)f2bf(v[2]) << 32)
                         | ((unsigned long long)f2bf(v[3]) << 48);
    *reinterpret_cast<unsigned long long*>(Xb + i) = p;
  }
}

// ============== pre-pass: W [K][N] f32 -> WT [N][K] bf16 (transpose) ==========
__global__ __launch_bounds__(256)
void transpose_w(const float* __restrict__ W, unsigned short* __restrict__ WT,
                 int K, int N)
{
  __shared__ unsigned short t[32][33];
  const int tid = threadIdx.x;
  const int c   = tid & 31;
  const int r0  = tid >> 5;
  const int kt  = blockIdx.y * 32;
  const int nt  = blockIdx.x * 32;
  #pragma unroll
  for (int i = 0; i < 4; i++) {
    int kk = r0 + i * 8;
    t[c][kk] = f2bf(W[(size_t)(kt + kk) * N + nt + c]);
  }
  __syncthreads();
  #pragma unroll
  for (int i = 0; i < 4; i++) {
    int nn = r0 + i * 8;
    WT[(size_t)(nt + nn) * K + kt + c] = t[nn][c];
  }
}

// ===================== LDS-free bf16 GEMM: out = (A @ BT^T + bias)*cscale =====
// A : [M][K] bf16       BT : [N][K] bf16 (pre-transposed)
// wave tile 64x64 (4x4 WMMA tiles); block 8 waves = 2(M) x 4(N) -> 128x256 tile
// OUT_MODE 0: f32 [M][N] -> outF    1: bf16 [B][N/64][S][64]    2: bf16 [B][N/64][64][S]
template<int OUT_MODE>
__global__ __launch_bounds__(256)
void gemm_bf16(const unsigned short* __restrict__ A,
               const unsigned short* __restrict__ BT,
               const float* __restrict__ bias, float cscale,
               unsigned short* __restrict__ outB, float* __restrict__ outF,
               int M, int N, int K, int S)
{
  const int tid  = threadIdx.x;
  const int wave = tid >> 5;
  const int lane = tid & 31;
  const int h    = lane >> 4;
  const int ln   = lane & 15;
  const int m0 = blockIdx.y * 128 + (wave >> 2) * 64;
  const int n0 = blockIdx.x * 256 + (wave & 3) * 64;

  v8f acc[4][4];
  #pragma unroll
  for (int i = 0; i < 4; i++)
    #pragma unroll
    for (int j = 0; j < 4; j++)
      #pragma unroll
      for (int r = 0; r < 8; r++) acc[i][j][r] = 0.0f;

  const unsigned short* arow[4];
  #pragma unroll
  for (int mi = 0; mi < 4; mi++) arow[mi] = A + (size_t)(m0 + mi * 16 + ln) * K;
  const unsigned short* brow[4];
  #pragma unroll
  for (int ni = 0; ni < 4; ni++) brow[ni] = BT + (size_t)(n0 + ni * 16 + ln) * K;

  for (int k0 = 0; k0 < K; k0 += 32) {
    Frag16 af[4];
    #pragma unroll
    for (int mi = 0; mi < 4; mi++) {
      const unsigned short* p = arow[mi] + k0;
      af[mi].u[0] = *reinterpret_cast<const v4u*>(p + 8 * h);       // K = e+8h
      af[mi].u[1] = *reinterpret_cast<const v4u*>(p + 16 + 8 * h);  // K = e+8+8h
    }
    #pragma unroll
    for (int ni = 0; ni < 4; ni++) {
      Frag16 bf;
      const unsigned short* q = brow[ni] + k0;                      // K = e+8h | e+8+8h
      bf.u[0] = *reinterpret_cast<const v4u*>(q + 8 * h);
      bf.u[1] = *reinterpret_cast<const v4u*>(q + 16 + 8 * h);
      #pragma unroll
      for (int mi = 0; mi < 4; mi++)
        acc[mi][ni] = __builtin_amdgcn_wmma_f32_16x16x32_bf16(
            false, af[mi].v, false, bf.v, (short)0, acc[mi][ni], false, false);
    }
  }

  const int HT = N >> 6;
  #pragma unroll
  for (int mi = 0; mi < 4; mi++) {
    #pragma unroll
    for (int ni = 0; ni < 4; ni++) {
      #pragma unroll
      for (int r = 0; r < 8; r++) {
        int m = m0 + mi * 16 + r + 8 * h;
        int n = n0 + ni * 16 + ln;
        float val = (acc[mi][ni][r] + bias[n]) * cscale;
        if (OUT_MODE == 0) {
          outF[(size_t)m * N + n] = val;
        } else {
          int b  = m / S;
          int s  = m - b * S;
          int hd = n >> 6;
          int d  = n & 63;
          size_t o;
          if (OUT_MODE == 1) o = (((size_t)(b * HT + hd) * S) + s) * 64 + d;
          else               o = (((size_t)(b * HT + hd) * 64) + d) * S + s;
          outB[o] = f2bf(val);
        }
      }
    }
  }
}

// ============================ flash-style GQA attention =======================
// One wave owns 32 query rows (2 q-tiles) of one (b,h); K/V fragments shared
// across both q-tiles. 1/sqrt(dk) is pre-folded into Q.
#define PLD 40   // P staging row stride (ushorts); 80B rows, 16B aligned

__global__ __launch_bounds__(256)
void attn_kernel(const unsigned short* __restrict__ Qb,   // [B][H][S][64] (pre-scaled)
                 const unsigned short* __restrict__ Kb,   // [B][G][S][64]
                 const unsigned short* __restrict__ Vt,   // [B][G][64][S]
                 unsigned short* __restrict__ Ctx,        // [B][S][H*64] bf16
                 int Bn, int Hn, int Gn, int S)
{
  __shared__ __align__(16) unsigned short pl[8 * 32 * PLD];

  const int wIn  = threadIdx.x >> 5;
  const int lane = threadIdx.x & 31;
  const int h    = lane >> 4;
  const int ln   = lane & 15;
  const int wid  = blockIdx.x * 8 + wIn;
  const int qTiles = S >> 5;                    // 32-row q blocks
  const int total  = Bn * Hn * qTiles;
  if (wid >= total) return;

  const int qt = wid % qTiles;
  const int hd = (wid / qTiles) % Hn;
  const int b  = wid / (qTiles * Hn);
  const int g  = hd / (Hn / Gn);

  const unsigned short* Qp = Qb + (((size_t)(b * Hn + hd) * S) + (size_t)qt * 32) * 64;
  const unsigned short* Kp = Kb + ((size_t)(b * Gn + g) * S) * 64;
  const unsigned short* Vp = Vt + ((size_t)(b * Gn + g) * 64) * S;
  unsigned short* plw = &pl[wIn * 32 * PLD];

  Frag16 qa[2][2];
  #pragma unroll
  for (int qi = 0; qi < 2; qi++)
    #pragma unroll
    for (int dc = 0; dc < 2; dc++) {
      const unsigned short* qp = Qp + (qi * 16 + ln) * 64 + dc * 32;
      qa[qi][dc].u[0] = *reinterpret_cast<const v4u*>(qp + 8 * h);
      qa[qi][dc].u[1] = *reinterpret_cast<const v4u*>(qp + 16 + 8 * h);
    }

  v8f   o[2][4];
  float mrow[2][8], lrow[2][8];
  #pragma unroll
  for (int qi = 0; qi < 2; qi++) {
    #pragma unroll
    for (int t = 0; t < 4; t++)
      #pragma unroll
      for (int r = 0; r < 8; r++) o[qi][t][r] = 0.0f;
    #pragma unroll
    for (int r = 0; r < 8; r++) { mrow[qi][r] = -INFINITY; lrow[qi][r] = 0.0f; }
  }

  for (int kc = 0; kc < S; kc += 32) {
    // ---- K fragments, shared by both q-tiles
    Frag16 kb[2][2];
    #pragma unroll
    for (int nt = 0; nt < 2; nt++)
      #pragma unroll
      for (int dc = 0; dc < 2; dc++) {
        const unsigned short* kp = Kp + (size_t)(kc + nt * 16 + ln) * 64 + dc * 32;
        kb[nt][dc].u[0] = *reinterpret_cast<const v4u*>(kp + 8 * h);
        kb[nt][dc].u[1] = *reinterpret_cast<const v4u*>(kp + 16 + 8 * h);
      }

    // ---- scores (Q pre-scaled by 1/8)
    v8f sfr[2][2];
    #pragma unroll
    for (int qi = 0; qi < 2; qi++)
      #pragma unroll
      for (int nt = 0; nt < 2; nt++) {
        v8f s;
        #pragma unroll
        for (int r = 0; r < 8; r++) s[r] = 0.0f;
        s = __builtin_amdgcn_wmma_f32_16x16x32_bf16(
                false, qa[qi][0].v, false, kb[nt][0].v, (short)0, s, false, false);
        s = __builtin_amdgcn_wmma_f32_16x16x32_bf16(
                false, qa[qi][1].v, false, kb[nt][1].v, (short)0, s, false, false);
        sfr[qi][nt] = s;
      }

    // ---- online softmax (DPP cross-lane reduce, no LDS)
    #pragma unroll
    for (int qi = 0; qi < 2; qi++) {
      #pragma unroll
      for (int r = 0; r < 8; r++) {
        float c   = dpp_max16(fmaxf(sfr[qi][0][r], sfr[qi][1][r]));
        float nm  = fmaxf(mrow[qi][r], c);
        float fac = __expf(mrow[qi][r] - nm);
        mrow[qi][r] = nm;
        lrow[qi][r] *= fac;
        #pragma unroll
        for (int t = 0; t < 4; t++) o[qi][t][r] *= fac;
        float p0 = __expf(sfr[qi][0][r] - nm);
        float p1 = __expf(sfr[qi][1][r] - nm);
        lrow[qi][r] += p0 + p1;
        plw[(qi * 16 + r + 8 * h) * PLD + ln]      = f2bf(p0);
        plw[(qi * 16 + r + 8 * h) * PLD + 16 + ln] = f2bf(p1);
      }
    }

    // ---- P as A fragments (same-wave LDS store->load)
    Frag16 pa[2];
    #pragma unroll
    for (int qi = 0; qi < 2; qi++) {
      const unsigned short* pp = &plw[(qi * 16 + ln) * PLD];
      pa[qi].u[0] = *reinterpret_cast<const v4u*>(pp + 8 * h);
      pa[qi].u[1] = *reinterpret_cast<const v4u*>(pp + 16 + 8 * h);
    }

    // ---- context: V fragments shared by both q-tiles
    #pragma unroll
    for (int t = 0; t < 4; t++) {
      Frag16 vb;
      const unsigned short* vp = Vp + (size_t)(t * 16 + ln) * S + kc;
      vb.u[0] = *reinterpret_cast<const v4u*>(vp + 8 * h);
      vb.u[1] = *reinterpret_cast<const v4u*>(vp + 16 + 8 * h);
      #pragma unroll
      for (int qi = 0; qi < 2; qi++)
        o[qi][t] = __builtin_amdgcn_wmma_f32_16x16x32_bf16(
                       false, pa[qi].v, false, vb.v, (short)0, o[qi][t], false, false);
    }
  }

  const int DM = Hn * 64;
  #pragma unroll
  for (int qi = 0; qi < 2; qi++) {
    float invl[8];
    #pragma unroll
    for (int r = 0; r < 8; r++) invl[r] = 1.0f / dpp_sum16(lrow[qi][r]);
    #pragma unroll
    for (int t = 0; t < 4; t++) {
      #pragma unroll
      for (int r = 0; r < 8; r++) {
        int srow = qt * 32 + qi * 16 + r + 8 * h;
        int col  = hd * 64 + t * 16 + ln;
        Ctx[(size_t)(b * S + srow) * DM + col] = f2bf(o[qi][t][r] * invl[r]);
      }
    }
  }
}

// ================================== launcher =================================
extern "C" void kernel_launch(void* const* d_in, const int* in_sizes, int n_in,
                              void* d_out, int out_size, void* d_ws, size_t ws_size,
                              hipStream_t stream)
{
  const int B = 2, S = 2048, DM = 2048, H = 32, G = 8, KV = 512;
  const float* x  = (const float*)d_in[0];
  const float* wq = (const float*)d_in[1];
  const float* bq = (const float*)d_in[2];
  const float* wk = (const float*)d_in[3];
  const float* bk = (const float*)d_in[4];
  const float* wv = (const float*)d_in[5];
  const float* bv = (const float*)d_in[6];
  const float* wo = (const float*)d_in[7];
  const float* bo = (const float*)d_in[8];
  float* out = (float*)d_out;

  const size_t M = (size_t)B * S;                 // 4096
  unsigned short* ws  = (unsigned short*)d_ws;
  unsigned short* Xb  = ws;                       // M*DM
  unsigned short* WqT = Xb  + M * DM;             // DM*DM
  unsigned short* WkT = WqT + (size_t)DM * DM;    // KV*DM
  unsigned short* WvT = WkT + (size_t)KV * DM;    // KV*DM
  unsigned short* WoT = WvT + (size_t)KV * DM;    // DM*DM
  unsigned short* Qb  = WoT + (size_t)DM * DM;    // M*DM : [B][H][S][64]
  unsigned short* Kb  = Qb  + M * DM;             // M*KV : [B][G][S][64]
  unsigned short* Vt  = Kb  + M * KV;             // M*KV : [B][G][64][S]
  unsigned short* Ctx = Vt  + M * KV;             // M*DM : [B][S][2048]

  dim3 blk(256);

  conv_bf16<<<dim3((int)((M * DM) / (4 * 256))), blk, 0, stream>>>(x, Xb, (long)(M * DM));
  transpose_w<<<dim3(DM / 32, DM / 32), blk, 0, stream>>>(wq, WqT, DM, DM);
  transpose_w<<<dim3(KV / 32, DM / 32), blk, 0, stream>>>(wk, WkT, DM, KV);
  transpose_w<<<dim3(KV / 32, DM / 32), blk, 0, stream>>>(wv, WvT, DM, KV);
  transpose_w<<<dim3(DM / 32, DM / 32), blk, 0, stream>>>(wo, WoT, DM, DM);

  // projections; Q gets 1/sqrt(64) folded in
  gemm_bf16<1><<<dim3(DM / 256, (int)M / 128), blk, 0, stream>>>(
      Xb, WqT, bq, 0.125f, Qb, nullptr, (int)M, DM, DM, S);
  gemm_bf16<1><<<dim3(KV / 256, (int)M / 128), blk, 0, stream>>>(
      Xb, WkT, bk, 1.0f, Kb, nullptr, (int)M, KV, DM, S);
  gemm_bf16<2><<<dim3(KV / 256, (int)M / 128), blk, 0, stream>>>(
      Xb, WvT, bv, 1.0f, Vt, nullptr, (int)M, KV, DM, S);

  // attention: one wave per 32 query rows per (b,h)
  int totalWaves = B * H * (S / 32);              // 4096
  attn_kernel<<<dim3(totalWaves / 8), blk, 0, stream>>>(Qb, Kb, Vt, Ctx, B, H, G, S);

  // output projection -> f32 d_out
  gemm_bf16<0><<<dim3(DM / 256, (int)M / 128), blk, 0, stream>>>(
      Ctx, WoT, bo, 1.0f, nullptr, out, (int)M, DM, DM, S);
}
